// CGConvSortPool_19121194402166
// MI455X (gfx1250) — matrix operand order repaired and checked by reading, not compiled, not run on hardware
//
#include <hip/hip_runtime.h>

typedef _Float16 v16h __attribute__((ext_vector_type(16)));
typedef _Float16 v8h  __attribute__((ext_vector_type(8)));
typedef float    v8f  __attribute__((ext_vector_type(8)));

#define NF   26          // node feature dim F
#define DEA  52          // edge attr dim
#define ZD   104         // gate input dim 2F+DE
#define KPAD 128         // padded K for WMMA
#define NPG  512         // nodes per graph
#define NG   256         // graphs
#define NN   (NG*NPG)    // total nodes
#define NE   1048576     // edges
#define TOPK 128
#define LD   53          // latent dim (26+26+1)
#define DH   512         // dense hidden
#define FLT  1920        // flatten dim (32*60)

#define LOG2E 1.4426950408889634f
#define LN2   0.6931471805599453f

__device__ __forceinline__ unsigned int packh2(float a, float b) {
  union { _Float16 h[2]; unsigned int u; } p;
  p.h[0] = (_Float16)a; p.h[1] = (_Float16)b;
  return p.u;
}

// ---------------------------------------------------------------------------
// CGConv edge-message kernel: WMMA f16 16x16x32, f32 accumulate.
// Per wave: tile of 16 edges. z = [x[dst](26) | x[src](26) | edge_attr(52) | pad]
// msg = sigmoid(z Wf^T + bf) * softplus(z Ws^T + bs), atomically added to acc[dst].
// ---------------------------------------------------------------------------
__launch_bounds__(256)
__global__ void edge_msg_kernel(const float* __restrict__ xin,
                                const float* __restrict__ eattr,
                                const int*   __restrict__ esrc,
                                const int*   __restrict__ edst,
                                const float* __restrict__ Wf,
                                const float* __restrict__ bf,
                                const float* __restrict__ Ws,
                                const float* __restrict__ bs,
                                float* __restrict__ acc) {
  __shared__ __align__(32) _Float16 wpf[32 * KPAD];
  __shared__ __align__(32) _Float16 wps[32 * KPAD];
  __shared__ float bfl[32], bsl[32];
  __shared__ __align__(32) _Float16 zbuf[8][16 * KPAD];
  __shared__ int dbuf[8][16];

  const int tid = threadIdx.x;

  // Stage weights into LDS as f16, padded [32][128] (zeros outside 26x104).
  for (int i = tid; i < 32 * KPAD; i += 256) {
    int o = i / KPAD, k = i % KPAD;
    float vf = 0.f, vs = 0.f;
    if (o < NF && k < ZD) { vf = Wf[o * ZD + k]; vs = Ws[o * ZD + k]; }
    wpf[i] = (_Float16)vf;
    wps[i] = (_Float16)vs;
  }
  if (tid < 32) {
    bfl[tid] = (tid < NF) ? bf[tid] : 0.f;
    bsl[tid] = (tid < NF) ? bs[tid] : 0.f;
  }
  __syncthreads();

  const int w    = tid >> 5;
  const int lane = tid & 31;
  const int lrow = lane & 15;
  const int lhi  = lane >> 4;

  // B fragments (32x16 f16): lanes 0-15 hold K 0..15, lanes 16-31 hold K 16..31.
  // Column n = lane&15 -> out channel o = n + 16t. Constant across tiles.
  v16h Bf[2][4], Bs[2][4];
  {
    const int kgrp = lhi << 4;
#pragma unroll
    for (int t = 0; t < 2; ++t) {
#pragma unroll
      for (int c = 0; c < 4; ++c) {
        Bf[t][c] = *(const v16h*)&wpf[(lrow + 16*t) * KPAD + 32*c + kgrp];
        Bs[t][c] = *(const v16h*)&wps[(lrow + 16*t) * KPAD + 32*c + kgrp];
      }
    }
  }

  const int slot = blockIdx.x * 8 + w;   // 1024 blocks * 8 waves = 8192 slots
  for (int it = 0; it < 8; ++it) {       // 8192 * 8 = 65536 tiles = E/16
    const int tile = slot + it * 8192;
    const int e    = tile * 16 + lrow;

    // ---- gather z tile into wave-private LDS (f16), straight-line loads.
    // lanes 0-15 fill K 0..63 of their edge row; lanes 16-31 fill K 64..127.
    unsigned int* zp = (unsigned int*)&zbuf[w][lrow * KPAD + lhi * 64];
    if (lhi == 0) {
      int si = esrc[e];
      int di = edst[e];
      dbuf[w][lrow] = di * NF;           // pre-scaled row offset for the atomics
      const float* xd = xin + di * NF;
      const float* xs = xin + si * NF;
      const float* ep = eattr + (size_t)e * DEA;
      float v[64];
#pragma unroll
      for (int j = 0; j < NF; ++j) v[j] = xd[j];
#pragma unroll
      for (int j = 0; j < NF; ++j) v[NF + j] = xs[j];
#pragma unroll
      for (int j = 0; j < 12; ++j) v[2*NF + j] = ep[j];
#pragma unroll
      for (int j = 0; j < 32; ++j) zp[j] = packh2(v[2*j], v[2*j + 1]);
    } else {
      const float* ep = eattr + (size_t)e * DEA + 12;
      float v[40];
#pragma unroll
      for (int j = 0; j < 40; ++j) v[j] = ep[j];
#pragma unroll
      for (int j = 0; j < 20; ++j) zp[j] = packh2(v[2*j], v[2*j + 1]);
#pragma unroll
      for (int j = 20; j < 32; ++j) zp[j] = 0u;
    }
    // Wave-private LDS region + in-order DS pipeline: compiler fence suffices.
    asm volatile("" ::: "memory");

    // ---- WMMA: 4 K-chunks x 2 N-tiles x 2 matrices
    v8f cf0 = {}, cf1 = {}, cs0 = {}, cs1 = {};
    const _Float16* zr = &zbuf[w][lrow * KPAD + lhi * 16];
#pragma unroll
    for (int c = 0; c < 4; ++c) {
      v8h lo = *(const v8h*)(zr + 32*c);
      v8h hi = *(const v8h*)(zr + 32*c + 8);
      v16h a;
#pragma unroll
      for (int j = 0; j < 8; ++j) { a[j] = lo[j]; a[j + 8] = hi[j]; }
      cf0 = __builtin_amdgcn_wmma_f32_16x16x32_f16(false, a, false, Bf[0][c], (short)0, cf0, false, false);
      cf1 = __builtin_amdgcn_wmma_f32_16x16x32_f16(false, a, false, Bf[1][c], (short)0, cf1, false, false);
      cs0 = __builtin_amdgcn_wmma_f32_16x16x32_f16(false, a, false, Bs[0][c], (short)0, cs0, false, false);
      cs1 = __builtin_amdgcn_wmma_f32_16x16x32_f16(false, a, false, Bs[1][c], (short)0, cs1, false, false);
    }

    // ---- epilogue: sigmoid * softplus via raw v_exp/v_log (base-2), scatter-add
    const int m0 = lhi * 8;     // D layout: VGPR r -> M = r (+8 for upper lanes)
    int dloc[8];
#pragma unroll
    for (int r = 0; r < 8; ++r) dloc[r] = dbuf[w][m0 + r];
#pragma unroll
    for (int t = 0; t < 2; ++t) {
      const int o   = lrow + 16*t;
      const bool vo = (o < NF);
      const float bfv = bfl[o], bsv = bsl[o];
      const v8f& gf = t ? cf1 : cf0;
      const v8f& gs = t ? cs1 : cs0;
#pragma unroll
      for (int r = 0; r < 8; ++r) {
        float gv = gf[r] + bfv;
        float sv = gs[r] + bsv;
        float sg = __builtin_amdgcn_rcpf(1.f + __builtin_amdgcn_exp2f(-gv * LOG2E));
        float ex = __builtin_amdgcn_exp2f(-fabsf(sv) * LOG2E);
        float sp = fmaxf(sv, 0.f) + LN2 * __builtin_amdgcn_logf(1.f + ex);
        float msg = sg * sp;
        if (vo) atomicAdd(&acc[dloc[r] + o], msg);
      }
    }
    asm volatile("" ::: "memory");   // order epilogue LDS reads vs next gather
  }
}

// ---------------------------------------------------------------------------
__global__ void node_update(const float* __restrict__ xin,
                            const float* __restrict__ acc,
                            float* __restrict__ hout, int n) {
  int i = blockIdx.x * 256 + threadIdx.x;
  if (i < n) hout[i] = tanhf(xin[i] + acc[i]);
}

__launch_bounds__(256)
__global__ void softmax_kernel(const float* __restrict__ ic, float* __restrict__ s) {
  __shared__ float red[256];
  int g = blockIdx.x, tid = threadIdx.x;
  const float* p = ic + g * NPG;
  float v0 = p[tid], v1 = p[tid + 256];
  red[tid] = fmaxf(v0, v1);
  __syncthreads();
  for (int st = 128; st > 0; st >>= 1) {
    if (tid < st) red[tid] = fmaxf(red[tid], red[tid + st]);
    __syncthreads();
  }
  float mx = red[0];
  __syncthreads();
  float e0 = __expf(v0 - mx), e1 = __expf(v1 - mx);
  red[tid] = e0 + e1;
  __syncthreads();
  for (int st = 128; st > 0; st >>= 1) {
    if (tid < st) red[tid] += red[tid + st];
    __syncthreads();
  }
  float inv = 1.f / red[0];
  s[g * NPG + tid]       = e0 * inv;
  s[g * NPG + tid + 256] = e1 * inv;
}

__global__ void xcat_kernel(const float* __restrict__ h1, const float* __restrict__ h2,
                            const float* __restrict__ s, float* __restrict__ xc) {
  int i = blockIdx.x * 256 + threadIdx.x;
  if (i >= NN * LD) return;
  int n = i / LD, l = i % LD;
  float v = (l < NF) ? h1[n * NF + l] : (l < 2*NF) ? h2[n * NF + (l - NF)] : s[n];
  xc[i] = v;
}

// Per-graph stable descending bitonic sort of 512 keys, gather top-128 rows.
__launch_bounds__(256)
__global__ void sortpool_kernel(const float* __restrict__ s,
                                const float* __restrict__ xc,
                                float* __restrict__ pooled) {
  __shared__ float key[512];
  __shared__ int   idx[512];
  int g = blockIdx.x, tid = threadIdx.x;
  for (int j = tid; j < 512; j += 256) { key[j] = s[g * NPG + j]; idx[j] = j; }
  __syncthreads();
  for (int k = 2; k <= 512; k <<= 1) {
    for (int j = k >> 1; j > 0; j >>= 1) {
      for (int t = tid; t < 512; t += 256) {
        int ixj = t ^ j;
        if (ixj > t) {
          bool up = ((t & k) == 0);
          float ka = key[t], kb = key[ixj];
          int   ia = idx[t], ib = idx[ixj];
          bool before = (ka > kb) || (ka == kb && ia < ib);  // stable descending
          if (up ? !before : before) {
            key[t] = kb; key[ixj] = ka;
            idx[t] = ib; idx[ixj] = ia;
          }
        }
      }
      __syncthreads();
    }
  }
  for (int i = tid; i < TOPK * LD; i += 256) {
    int k2 = i / LD, l = i % LD;
    pooled[g * TOPK * LD + i] = xc[(g * NPG + idx[k2]) * LD + l];
  }
}

// conv1d(1->16, kernel=stride=53) == per-pooled-node linear, + relu + maxpool(2,2)
__launch_bounds__(256)
__global__ void conv1_pool_kernel(const float* __restrict__ pooled,
                                  const float* __restrict__ Wc1,
                                  const float* __restrict__ bc1,
                                  float* __restrict__ hmax) {
  __shared__ float wl[16 * LD];
  __shared__ float bl[16];
  int g = blockIdx.x, tid = threadIdx.x;
  for (int i = tid; i < 16 * LD; i += 256) wl[i] = Wc1[i];
  if (tid < 16) bl[tid] = bc1[tid];
  __syncthreads();
  const float* pg = pooled + g * TOPK * LD;
  for (int p = tid; p < 16 * 64; p += 256) {
    int o = p / 64, m = p % 64;
    float best = 0.f;                      // relu outputs are >= 0
    for (int h = 0; h < 2; ++h) {
      const float* row = pg + (2*m + h) * LD;
      float a = bl[o];
      for (int l = 0; l < LD; ++l) a += row[l] * wl[o * LD + l];
      best = fmaxf(best, fmaxf(a, 0.f));
    }
    hmax[g * 1024 + p] = best;
  }
}

// conv1d(16->32, kernel=5) + relu -> [G, 32, 60] flattened, stored f16 for GEMM
__global__ void conv2_kernel(const float* __restrict__ hmax,
                             const float* __restrict__ Wc2,
                             const float* __restrict__ bc2,
                             _Float16* __restrict__ f2h) {
  int id = blockIdx.x * 256 + threadIdx.x;
  if (id >= NG * FLT) return;
  int g = id / FLT, r = id % FLT, o = r / 60, jj = r % 60;
  const float* hm = hmax + g * 1024;
  float a = bc2[o];
  for (int i = 0; i < 16; ++i) {
    const float* wrow = Wc2 + o * 80 + i * 5;
    const float* hrow = hm + i * 64 + jj;
    for (int t = 0; t < 5; ++t) a += hrow[t] * wrow[t];
  }
  f2h[id] = (_Float16)fmaxf(a, 0.f);
}

// ---------------------------------------------------------------------------
// Generic WMMA GEMM:  out[M x N] = relu(A[M x K](f16) * W[N x K]^T(f32) + bias)
// One 16x16 output tile per wave; K multiple of 32; grid*8 waves == tile count.
// ---------------------------------------------------------------------------
__launch_bounds__(256)
__global__ void gemm_relu_kernel(const _Float16* __restrict__ A,
                                 const float* __restrict__ W,
                                 const float* __restrict__ bias,
                                 _Float16* __restrict__ outH,
                                 float* __restrict__ outF,
                                 int N, int K) {
  const int tid = threadIdx.x;
  const int wv = tid >> 5, lane = tid & 31;
  const int lrow = lane & 15, lhi = lane >> 4;
  const int ntiles = N >> 4;
  const int tile = blockIdx.x * 8 + wv;
  const int mt = tile / ntiles, nt = tile % ntiles;

  const _Float16* ap = A + (size_t)(mt * 16 + lrow) * K + lhi * 16;
  const float*    wp = W + (size_t)(nt * 16 + lrow) * K + (lhi << 4);

  v8f c = {};
  const int chunks = K >> 5;
  for (int ci = 0; ci < chunks; ++ci) {
    v8h lo = *(const v8h*)ap;
    v8h hi = *(const v8h*)(ap + 8);
    v16h a;
#pragma unroll
    for (int j = 0; j < 8; ++j) { a[j] = lo[j]; a[j + 8] = hi[j]; }
    float4 w0 = *(const float4*)(wp);
    float4 w1 = *(const float4*)(wp + 4);
    float4 w2 = *(const float4*)(wp + 8);
    float4 w3 = *(const float4*)(wp + 12);
    v16h b;
    b[0]  = (_Float16)w0.x; b[1]  = (_Float16)w0.y; b[2]  = (_Float16)w0.z; b[3]  = (_Float16)w0.w;
    b[4]  = (_Float16)w1.x; b[5]  = (_Float16)w1.y; b[6]  = (_Float16)w1.z; b[7]  = (_Float16)w1.w;
    b[8]  = (_Float16)w2.x; b[9]  = (_Float16)w2.y; b[10] = (_Float16)w2.z; b[11] = (_Float16)w2.w;
    b[12] = (_Float16)w3.x; b[13] = (_Float16)w3.y; b[14] = (_Float16)w3.z; b[15] = (_Float16)w3.w;
    c = __builtin_amdgcn_wmma_f32_16x16x32_f16(false, a, false, b, (short)0, c, false, false);
    ap += 32;
    wp += 32;
  }

  const int n_out = nt * 16 + lrow;
  const float bv = bias[n_out];
  const int mbase = mt * 16 + (lhi ? 8 : 0);
#pragma unroll
  for (int r = 0; r < 8; ++r) {
    float v = fmaxf(c[r] + bv, 0.f);
    int m = mbase + r;
    if (outH) outH[(size_t)m * N + n_out] = (_Float16)v;
    if (outF) outF[(size_t)m * N + n_out] = v;
  }
}

// Final 256 -> 1 dot per graph
__launch_bounds__(256)
__global__ void final_kernel(const float* __restrict__ h1f,
                             const float* __restrict__ W2,
                             const float* __restrict__ b2,
                             float* __restrict__ out) {
  __shared__ float red[256];
  int g = blockIdx.x, tid = threadIdx.x;
  red[tid] = h1f[g * 256 + tid] * W2[tid];
  __syncthreads();
  for (int st = 128; st > 0; st >>= 1) {
    if (tid < st) red[tid] += red[tid + st];
    __syncthreads();
  }
  if (tid == 0) out[g] = red[0] + b2[0];
}

// ---------------------------------------------------------------------------
extern "C" void kernel_launch(void* const* d_in, const int* in_sizes, int n_in,
                              void* d_out, int out_size, void* d_ws, size_t ws_size,
                              hipStream_t stream) {
  (void)in_sizes; (void)n_in; (void)out_size; (void)ws_size;
  const float* x   = (const float*)d_in[0];
  const float* ea  = (const float*)d_in[1];
  const float* ic  = (const float*)d_in[2];
  const float* Wf1 = (const float*)d_in[3];
  const float* bf1 = (const float*)d_in[4];
  const float* Ws1 = (const float*)d_in[5];
  const float* bs1 = (const float*)d_in[6];
  const float* Wf2 = (const float*)d_in[7];
  const float* bf2 = (const float*)d_in[8];
  const float* Ws2 = (const float*)d_in[9];
  const float* bs2 = (const float*)d_in[10];
  const float* Wc1 = (const float*)d_in[11];
  const float* bc1 = (const float*)d_in[12];
  const float* Wc2 = (const float*)d_in[13];
  const float* bc2 = (const float*)d_in[14];
  const float* W0  = (const float*)d_in[15];
  const float* b0  = (const float*)d_in[16];
  const float* W1  = (const float*)d_in[17];
  const float* b1  = (const float*)d_in[18];
  const float* W2  = (const float*)d_in[19];
  const float* b2  = (const float*)d_in[20];
  const int*   ei  = (const int*)d_in[21];
  const int* esrc = ei;            // edge_index[0] = src
  const int* edst = ei + NE;       // edge_index[1] = dst
  float* out = (float*)d_out;

  char* base = (char*)d_ws;
  auto alloc = [&](size_t bytes) {
    void* r = (void*)base;
    base += (bytes + 255) & ~(size_t)255;
    return r;
  };
  float*    acc    = (float*)alloc((size_t)NN * NF * 4);
  float*    h1     = (float*)alloc((size_t)NN * NF * 4);
  float*    h2     = (float*)alloc((size_t)NN * NF * 4);
  float*    sgm    = (float*)alloc((size_t)NN * 4);
  float*    xc     = (float*)alloc((size_t)NN * LD * 4);
  float*    pooled = (float*)alloc((size_t)NG * TOPK * LD * 4);
  float*    hmax   = (float*)alloc((size_t)NG * 1024 * 4);
  _Float16* f2h    = (_Float16*)alloc((size_t)NG * FLT * 2);
  _Float16* h0h    = (_Float16*)alloc((size_t)NG * DH * 2);
  float*    h1f    = (float*)alloc((size_t)NG * 256 * 4);

  hipMemsetAsync(acc, 0, sizeof(float) * (size_t)NN * NF, stream);
  edge_msg_kernel<<<1024, 256, 0, stream>>>(x, ea, esrc, edst, Wf1, bf1, Ws1, bs1, acc);
  node_update<<<(NN * NF + 255) / 256, 256, 0, stream>>>(x, acc, h1, NN * NF);

  hipMemsetAsync(acc, 0, sizeof(float) * (size_t)NN * NF, stream);
  edge_msg_kernel<<<1024, 256, 0, stream>>>(h1, ea, esrc, edst, Wf2, bf2, Ws2, bs2, acc);
  node_update<<<(NN * NF + 255) / 256, 256, 0, stream>>>(h1, acc, h2, NN * NF);

  softmax_kernel<<<NG, 256, 0, stream>>>(ic, sgm);
  xcat_kernel<<<(NN * LD + 255) / 256, 256, 0, stream>>>(h1, h2, sgm, xc);
  sortpool_kernel<<<NG, 256, 0, stream>>>(sgm, xc, pooled);
  conv1_pool_kernel<<<NG, 256, 0, stream>>>(pooled, Wc1, bc1, hmax);
  conv2_kernel<<<(NG * FLT + 255) / 256, 256, 0, stream>>>(hmax, Wc2, bc2, f2h);

  // MLP head: 1920 -> 512 (WMMA) -> 256 (WMMA) -> 1
  gemm_relu_kernel<<<64, 256, 0, stream>>>(f2h, W0, b0, h0h, (float*)nullptr, DH, FLT);
  gemm_relu_kernel<<<32, 256, 0, stream>>>(h0h, W1, b1, (_Float16*)nullptr, h1f, 256, DH);
  final_kernel<<<NG, 256, 0, stream>>>(h1f, W2, b2, out);
}
